// FC_44985487458808
// MI455X (gfx1250) — compile-verified, matrix operand order checked by reading, not run on hardware
//
#include <hip/hip_runtime.h>

// ---------------- types for WMMA ----------------
typedef __attribute__((ext_vector_type(16))) __bf16 v16bf;
typedef __attribute__((ext_vector_type(8)))  float  v8f;
typedef __attribute__((ext_vector_type(4)))  unsigned int v4u;
typedef __attribute__((ext_vector_type(4)))  int v4i;

union Frag { v16bf v; v4u q[2]; };

#if defined(__gfx1250__) && __has_builtin(__builtin_amdgcn_global_load_async_to_lds_b128)
#define HAVE_ASYNC_LDS 1
#else
#define HAVE_ASYNC_LDS 0
#endif

__device__ __forceinline__ void wait_async_lds() {
#if HAVE_ASYNC_LDS
#if __has_builtin(__builtin_amdgcn_s_wait_asynccnt)
  __builtin_amdgcn_s_wait_asynccnt(0);
#else
  asm volatile("s_wait_asynccnt 0x0" ::: "memory");
#endif
#endif
}

__device__ __forceinline__ unsigned short f2bf(float f) {
  unsigned int u = __float_as_uint(f);
  u += 0x7FFFu + ((u >> 16) & 1u);      // round-to-nearest-even
  return (unsigned short)(u >> 16);
}

// ---------------- conversion kernels ----------------
// X [M,K] fp32 row-major -> Y [Mp,Kp] bf16 row-major, zero padded
__global__ void k_cvt_bf16(const float* __restrict__ X, unsigned short* __restrict__ Y,
                           int M, int K, int Kp, long long total) {
  long long i = (long long)blockIdx.x * blockDim.x + threadIdx.x;
  if (i >= total) return;
  int r = (int)(i / Kp), c = (int)(i % Kp);
  float v = (r < M && c < K) ? X[(size_t)r * K + c] : 0.0f;
  Y[i] = f2bf(v);
}

// B [K,N] fp32 row-major -> Y [Np,Kp] bf16 (i.e. B^T), zero padded
__global__ void k_cvt_bf16_T(const float* __restrict__ B, unsigned short* __restrict__ Y,
                             int K, int N, int Kp, long long total) {
  long long i = (long long)blockIdx.x * blockDim.x + threadIdx.x;
  if (i >= total) return;
  int n = (int)(i / Kp), k = (int)(i % Kp);
  float v = (n < N && k < K) ? B[(size_t)k * N + n] : 0.0f;
  Y[i] = f2bf(v);
}

// ---------------- WMMA GEMM ----------------
// C[M,N] = A[Mp,Kp](bf16) * BT[Np64,Kp]^T(bf16) (+bias).
// Block = 256 threads = 8 waves. Block tile: 128(M) x 64(N).
// Wave w -> M-tile (bm*8+w); all waves share a 64-column B panel staged in LDS
// (double buffered, async global->LDS when available). Each wave runs 4 WMMAs
// per K-step, reusing its A fragment 4x.
#define LDS_ROW 80                      // 64B data + 16B pad (bank-conflict free)
#define LDS_BUF (64 * LDS_ROW)          // 5120 B per buffer

__global__ __launch_bounds__(256) void k_wmma_gemm(
    const unsigned short* __restrict__ A,
    const unsigned short* __restrict__ BT,
    const float* __restrict__ bias,
    float* __restrict__ C,
    int M, int N, int Kp, int tilesM, int nBlocks) {
  __shared__ __attribute__((aligned(16))) unsigned char smem[2 * LDS_BUF];

  const int bm = blockIdx.x / nBlocks;
  const int bn = blockIdx.x - bm * nBlocks;
  const int w    = threadIdx.x >> 5;
  const int lane = threadIdx.x & 31;
  const int l15  = lane & 15;
  const int hi   = lane >> 4;            // 0/1: selects K sub-block of fragment

  int tm = bm * 8 + w;
  if (tm >= tilesM) tm = tilesM - 1;     // clamp: duplicate tile, identical data
  const unsigned short* ap = A + (size_t)(tm * 16 + l15) * Kp + (hi << 3);

  // staging role: 256 threads move 64 rows x 64B per K-step (one b128 each)
  const int srow = threadIdx.x >> 2;     // 0..63
  const int sseg = threadIdx.x & 3;      // 0..3
  const unsigned short* sp = BT + (size_t)(bn * 64 + srow) * Kp + sseg * 8;
  unsigned char* sdst = &smem[srow * LDS_ROW + sseg * 16];

  auto stage = [&](int buf, int k0) {
    const unsigned short* g = sp + k0;
    unsigned char* l = sdst + buf * LDS_BUF;
#if HAVE_ASYNC_LDS
    __builtin_amdgcn_global_load_async_to_lds_b128(
        (__attribute__((address_space(1))) v4i*)(void*)g,
        (__attribute__((address_space(3))) v4i*)(void*)l, 0, 0);
#else
    *(v4u*)l = *(const v4u*)g;
#endif
  };

  v8f z = {0.f, 0.f, 0.f, 0.f, 0.f, 0.f, 0.f, 0.f};
  v8f acc[4] = {z, z, z, z};

  const int nsteps = Kp >> 5;
  stage(0, 0);
  for (int i = 0; i < nsteps; ++i) {
    const int k0 = i << 5;
    wait_async_lds();
    __syncthreads();                      // buf[i&1] ready everywhere
    if (i + 1 < nsteps) {
      stage((i + 1) & 1, k0 + 32);
      __builtin_prefetch(ap + k0 + 32, 0, 1);   // global_prefetch_b8 (A stream)
    }
    Frag fa;
    fa.q[0] = *(const v4u*)(ap + k0);
    fa.q[1] = *(const v4u*)(ap + k0 + 16);
    const unsigned char* bb = &smem[(i & 1) * LDS_BUF];
#pragma unroll
    for (int f = 0; f < 4; ++f) {
      const unsigned char* p = bb + (f * 16 + l15) * LDS_ROW + hi * 16;
      Frag fb;
      fb.q[0] = *(const v4u*)(p);
      fb.q[1] = *(const v4u*)(p + 32);
      acc[f] = __builtin_amdgcn_wmma_f32_16x16x32_bf16(false, fa.v, false, fb.v,
                                                       (short)0, acc[f], false, false);
    }
    __syncthreads();                      // all reads of buf[i&1] done before restage
  }

  const int mbase = tm * 16 + (hi << 3);
#pragma unroll
  for (int f = 0; f < 4; ++f) {
    int n = bn * 64 + f * 16 + l15;
    if (n < N) {
      float bv = bias ? bias[n] : 0.0f;
#pragma unroll
      for (int r = 0; r < 8; ++r) {       // acc VGPR r -> M = mbase + r
        int m = mbase + r;
        if (m < M) C[(size_t)m * N + n] = acc[f][r] + bv;
      }
    }
  }
}

// ---------------- GCN helper kernels ----------------
__global__ void k_deg_accum(const int* __restrict__ ei, const float* __restrict__ ew,
                            float* __restrict__ deg, int E) {
  int e = blockIdx.x * blockDim.x + threadIdx.x;
  if (e < E) atomicAdd(&deg[ei[E + e]], ew[e]);  // dst row is ei[1][e]
}

__global__ void k_deg_finish(float* __restrict__ deg, int N) {
  int i = blockIdx.x * blockDim.x + threadIdx.x;
  if (i >= N) return;
  float d = deg[i] + 1.0f;                       // self-loop weight 1
  deg[i] = (d > 0.0f) ? rsqrtf(d) : 0.0f;        // store dinv in place
}

__global__ void k_edge_agg(const int* __restrict__ ei, const float* __restrict__ ew,
                           const float* __restrict__ dinv, const float* __restrict__ H,
                           float* __restrict__ G, int E, int F) {
  int e = blockIdx.x;
  if (e >= E) return;
  int src = ei[e];
  int dst = ei[E + e];
  float norm = dinv[src] * ew[e] * dinv[dst];
  const float* hs = H + (size_t)src * F;
  float* gd = G + (size_t)dst * F;
  for (int f = threadIdx.x; f < F; f += blockDim.x)
    atomicAdd(&gd[f], norm * hs[f]);
}

// self-loop term + bias + LeakyReLU, in place on G
__global__ void k_gcn_finish(const float* __restrict__ H, const float* __restrict__ dinv,
                             const float* __restrict__ bias, float* __restrict__ G,
                             int N, int F) {
  long long idx = (long long)blockIdx.x * blockDim.x + threadIdx.x;
  if (idx >= (long long)N * F) return;
  int i = (int)(idx / F), f = (int)(idx % F);
  float di = dinv[i];
  float v = G[idx] + di * di * H[idx] + bias[f];
  G[idx] = (v >= 0.0f) ? v : 0.01f * v;
}

__global__ void k_gather_rows(const float* __restrict__ G, const int* __restrict__ idxv,
                              float* __restrict__ feat, int B, int F, int ld, int colOff) {
  long long idx = (long long)blockIdx.x * blockDim.x + threadIdx.x;
  if (idx >= (long long)B * F) return;
  int b = (int)(idx / F), f = (int)(idx % F);
  feat[(size_t)b * ld + colOff + f] = G[(size_t)idxv[b] * F + f];
}

__global__ void k_copy_cols(const float* __restrict__ X, float* __restrict__ feat,
                            int B, int W, int ld, int colOff) {
  long long idx = (long long)blockIdx.x * blockDim.x + threadIdx.x;
  if (idx >= (long long)B * W) return;
  int b = (int)(idx / W), f = (int)(idx % W);
  feat[(size_t)b * ld + colOff + f] = X[(size_t)b * W + f];
}

// ---------------- BatchNorm (training-mode, biased var) ----------------
__global__ void k_bn_stats(const float* __restrict__ X, float* __restrict__ mean,
                           float* __restrict__ istd, int B, int N) {
  int j = blockIdx.x * blockDim.x + threadIdx.x;
  if (j >= N) return;
  float s = 0.f, s2 = 0.f;
  for (int i = 0; i < B; ++i) {
    float v = X[(size_t)i * N + j];
    s += v; s2 += v * v;
  }
  float m = s / (float)B;
  float var = fmaxf(s2 / (float)B - m * m, 0.f);
  mean[j] = m;
  istd[j] = rsqrtf(var + 1e-5f);
}

// mode 0: ReLU, mode 1: LeakyReLU(0.01). In place.
__global__ void k_bn_apply(float* __restrict__ X, const float* __restrict__ mean,
                           const float* __restrict__ istd, const float* __restrict__ g,
                           const float* __restrict__ beta, int B, int N, int mode) {
  long long idx = (long long)blockIdx.x * blockDim.x + threadIdx.x;
  if (idx >= (long long)B * N) return;
  int j = (int)(idx % N);
  float v = (X[idx] - mean[j]) * istd[j] * g[j] + beta[j];
  X[idx] = (mode == 0) ? fmaxf(v, 0.f) : ((v >= 0.f) ? v : 0.01f * v);
}

// ---------------- final head: y = Z[2048,64] @ W[64] + b ----------------
__global__ void k_head(const float* __restrict__ Z, const float* __restrict__ W,
                       const float* __restrict__ b, float* __restrict__ y, int B) {
  int i = blockIdx.x * blockDim.x + threadIdx.x;
  if (i >= B) return;
  float s = b[0];
#pragma unroll
  for (int f = 0; f < 64; ++f) s += Z[(size_t)i * 64 + f] * W[f];
  y[i] = s;
}

// ---------------- host helpers ----------------
static inline int cdiv_i(long long a, long long b) { return (int)((a + b - 1) / b); }

// Np64 must be a multiple of 64 (BT has Np64 rows)
static void launch_gemm(const unsigned short* A, const unsigned short* BT, const float* bias,
                        float* C, int M, int N, int Kp, int Mp, int Np64, hipStream_t s) {
  int tilesM = Mp / 16;
  int mBlocks = cdiv_i(tilesM, 8);
  int nBlocks = Np64 / 64;
  k_wmma_gemm<<<mBlocks * nBlocks, 256, 0, s>>>(A, BT, bias, C, M, N, Kp, tilesM, nBlocks);
}

static void run_gcn(const float* X, const int* ei, const float* ew, const float* W,
                    const float* bias, const int* gidx, float* outFeat, int colOff,
                    int Nnodes, int Mp, int K, int Kp, int E,
                    unsigned short* A16, unsigned short* BT16, float* H, float* G,
                    float* deg, hipStream_t s) {
  const int Fo = 1024;
  (void)hipMemsetAsync(deg, 0, (size_t)Nnodes * sizeof(float), s);
  k_deg_accum<<<cdiv_i(E, 256), 256, 0, s>>>(ei, ew, deg, E);
  k_deg_finish<<<cdiv_i(Nnodes, 256), 256, 0, s>>>(deg, Nnodes);
  long long ta = (long long)Mp * Kp;
  k_cvt_bf16<<<cdiv_i(ta, 256), 256, 0, s>>>(X, A16, Nnodes, K, Kp, ta);
  long long tb = (long long)Fo * Kp;
  k_cvt_bf16_T<<<cdiv_i(tb, 256), 256, 0, s>>>(W, BT16, K, Fo, Kp, tb);
  launch_gemm(A16, BT16, nullptr, H, Nnodes, Fo, Kp, Mp, Fo, s);
  (void)hipMemsetAsync(G, 0, (size_t)Nnodes * Fo * sizeof(float), s);
  k_edge_agg<<<E, 256, 0, s>>>(ei, ew, deg, H, G, E, Fo);
  k_gcn_finish<<<cdiv_i((long long)Nnodes * Fo, 256), 256, 0, s>>>(H, deg, bias, G, Nnodes, Fo);
  k_gather_rows<<<cdiv_i((long long)2048 * Fo, 256), 256, 0, s>>>(G, gidx, outFeat, 2048, Fo, 3372, colOff);
}

extern "C" void kernel_launch(void* const* d_in, const int* in_sizes, int n_in,
                              void* d_out, int out_size, void* d_ws, size_t ws_size,
                              hipStream_t stream) {
  (void)in_sizes; (void)n_in; (void)out_size; (void)ws_size;
  const int*   d_index = (const int*)d_in[0];
  const int*   p_index = (const int*)d_in[1];
  const float* d_vecs  = (const float*)d_in[2];
  const float* p_emb   = (const float*)d_in[3];
  const float* d_ecfps = (const float*)d_in[4];
  const int*   d_ei    = (const int*)d_in[5];
  const float* d_ew    = (const float*)d_in[6];
  const float* p_gos   = (const float*)d_in[7];
  const int*   p_ei    = (const int*)d_in[8];
  const float* p_ew    = (const float*)d_in[9];
  const float* d_gcn_W = (const float*)d_in[10];
  const float* d_gcn_b = (const float*)d_in[11];
  const float* p_gcn_W = (const float*)d_in[12];
  const float* p_gcn_b = (const float*)d_in[13];
  const float* enc_W1  = (const float*)d_in[14];
  const float* enc_b1  = (const float*)d_in[15];
  const float* enc_g1  = (const float*)d_in[16];
  const float* enc_be1 = (const float*)d_in[17];
  const float* enc_W2  = (const float*)d_in[18];
  const float* enc_b2  = (const float*)d_in[19];
  const float* enc_g2  = (const float*)d_in[20];
  const float* enc_be2 = (const float*)d_in[21];
  const float* dec_W1  = (const float*)d_in[22];
  const float* dec_b1  = (const float*)d_in[23];
  const float* dec_g1  = (const float*)d_in[24];
  const float* dec_be1 = (const float*)d_in[25];
  const float* dec_W2  = (const float*)d_in[26];
  const float* dec_b2  = (const float*)d_in[27];
  const float* dec_g2  = (const float*)d_in[28];
  const float* dec_be2 = (const float*)d_in[29];
  const float* out_W1  = (const float*)d_in[30];
  const float* out_b1  = (const float*)d_in[31];
  const float* out_g1  = (const float*)d_in[32];
  const float* out_be1 = (const float*)d_in[33];
  const float* out_W2  = (const float*)d_in[34];
  const float* out_b2  = (const float*)d_in[35];

  const int B = 2048, F = 3372;
  // d_out layout: y[2048], encoded[2048*256], decoded[2048*3372], feature[2048*3372]
  float* outY    = (float*)d_out;
  float* outEnc  = outY + 2048;
  float* outDec  = outEnc + (size_t)B * 256;
  float* outFeat = outDec + (size_t)B * F;

  // workspace layout (bytes): A16 32MB | BT16 8MB | H 44MB | G 44MB | small 1MB
  char* ws = (char*)d_ws;
  unsigned short* A16  = (unsigned short*)(ws);
  unsigned short* BT16 = (unsigned short*)(ws + (size_t)32 * 1024 * 1024);
  float* Hbuf = (float*)(ws + (size_t)40 * 1024 * 1024);
  float* Gbuf = (float*)(ws + (size_t)84 * 1024 * 1024);
  char*  sm   = ws + (size_t)128 * 1024 * 1024;
  float* deg  = (float*)(sm);                 // up to 10000 floats
  float* mean = (float*)(sm + 65536);         // up to 3372 floats
  float* istd = (float*)(sm + 131072);
  float* Zo   = (float*)(sm + 262144);        // 2048*64 floats

  hipStream_t s = stream;

  // ---- feature base columns ----
  k_copy_cols<<<cdiv_i((long long)B * 300, 256), 256, 0, s>>>(d_vecs, outFeat, B, 300, F, 0);
  k_copy_cols<<<cdiv_i((long long)B * 1024, 256), 256, 0, s>>>(p_emb, outFeat, B, 1024, F, 300);

  // ---- drug GCN: [10000,1024] x [1024,1024] -> feature cols 1324..2347 ----
  run_gcn(d_ecfps, d_ei, d_ew, d_gcn_W, d_gcn_b, d_index, outFeat, 1324,
          10000, 10000, 1024, 1024, 65536, A16, BT16, Hbuf, Gbuf, deg, s);

  // ---- protein GCN: [5000,2812] x [2812,1024] -> feature cols 2348..3371 ----
  run_gcn(p_gos, p_ei, p_ew, p_gcn_W, p_gcn_b, p_index, outFeat, 2348,
          5000, 5008, 2812, 2816, 25000, A16, BT16, Hbuf, Gbuf, deg, s);

  // ---- encoder layer 1: feature[2048,3372] @ enc_W1[3372,1024] ----
  {
    long long ta = (long long)2048 * 3392;
    k_cvt_bf16<<<cdiv_i(ta, 256), 256, 0, s>>>(outFeat, A16, 2048, 3372, 3392, ta);
    long long tb = (long long)1024 * 3392;
    k_cvt_bf16_T<<<cdiv_i(tb, 256), 256, 0, s>>>(enc_W1, BT16, 3372, 1024, 3392, tb);
    launch_gemm(A16, BT16, enc_b1, Hbuf, 2048, 1024, 3392, 2048, 1024, s);
    k_bn_stats<<<cdiv_i(1024, 256), 256, 0, s>>>(Hbuf, mean, istd, B, 1024);
    k_bn_apply<<<cdiv_i((long long)B * 1024, 256), 256, 0, s>>>(Hbuf, mean, istd, enc_g1, enc_be1, B, 1024, 0);
  }

  // ---- encoder layer 2 -> encoded [2048,256] in d_out ----
  {
    long long ta = (long long)2048 * 1024;
    k_cvt_bf16<<<cdiv_i(ta, 256), 256, 0, s>>>(Hbuf, A16, 2048, 1024, 1024, ta);
    long long tb = (long long)256 * 1024;
    k_cvt_bf16_T<<<cdiv_i(tb, 256), 256, 0, s>>>(enc_W2, BT16, 1024, 256, 1024, tb);
    launch_gemm(A16, BT16, enc_b2, outEnc, 2048, 256, 1024, 2048, 256, s);
    k_bn_stats<<<cdiv_i(256, 256), 256, 0, s>>>(outEnc, mean, istd, B, 256);
    k_bn_apply<<<cdiv_i((long long)B * 256, 256), 256, 0, s>>>(outEnc, mean, istd, enc_g2, enc_be2, B, 256, 0);
  }

  // ---- decoder layer 1: encoded @ dec_W1[256,1024] ----
  {
    long long ta = (long long)2048 * 256;
    k_cvt_bf16<<<cdiv_i(ta, 256), 256, 0, s>>>(outEnc, A16, 2048, 256, 256, ta);
    long long tb = (long long)1024 * 256;
    k_cvt_bf16_T<<<cdiv_i(tb, 256), 256, 0, s>>>(dec_W1, BT16, 256, 1024, 256, tb);
    launch_gemm(A16, BT16, dec_b1, Hbuf, 2048, 1024, 256, 2048, 1024, s);
    k_bn_stats<<<cdiv_i(1024, 256), 256, 0, s>>>(Hbuf, mean, istd, B, 1024);
    k_bn_apply<<<cdiv_i((long long)B * 1024, 256), 256, 0, s>>>(Hbuf, mean, istd, dec_g1, dec_be1, B, 1024, 0);
  }

  // ---- decoder layer 2 -> decoded [2048,3372] in d_out ----
  {
    long long ta = (long long)2048 * 1024;
    k_cvt_bf16<<<cdiv_i(ta, 256), 256, 0, s>>>(Hbuf, A16, 2048, 1024, 1024, ta);
    long long tb = (long long)3392 * 1024;
    k_cvt_bf16_T<<<cdiv_i(tb, 256), 256, 0, s>>>(dec_W2, BT16, 1024, 3372, 1024, tb);
    launch_gemm(A16, BT16, dec_b2, outDec, 2048, 3372, 1024, 2048, 3392, s);
    k_bn_stats<<<cdiv_i(3372, 256), 256, 0, s>>>(outDec, mean, istd, B, 3372);
    k_bn_apply<<<cdiv_i((long long)B * 3372, 256), 256, 0, s>>>(outDec, mean, istd, dec_g2, dec_be2, B, 3372, 0);
  }

  // ---- output head: encoded @ out_W1[256,64], BN+LeakyReLU, @ out_W2 + b ----
  {
    long long ta = (long long)2048 * 256;
    k_cvt_bf16<<<cdiv_i(ta, 256), 256, 0, s>>>(outEnc, A16, 2048, 256, 256, ta);
    long long tb = (long long)64 * 256;
    k_cvt_bf16_T<<<cdiv_i(tb, 256), 256, 0, s>>>(out_W1, BT16, 256, 64, 256, tb);
    launch_gemm(A16, BT16, out_b1, Zo, 2048, 64, 256, 2048, 64, s);
    k_bn_stats<<<cdiv_i(64, 256), 256, 0, s>>>(Zo, mean, istd, B, 64);
    k_bn_apply<<<cdiv_i((long long)B * 64, 256), 256, 0, s>>>(Zo, mean, istd, out_g1, out_be1, B, 64, 1);
    k_head<<<cdiv_i(B, 256), 256, 0, s>>>(Zo, out_W2, out_b2, outY, B);
  }
}
